// PersLoss_6828998001473
// MI455X (gfx1250) — matrix operand order repaired and checked by reading, not compiled, not run on hardware
//
#include <hip/hip_runtime.h>
#include <hip/hip_bf16.h>

typedef __attribute__((ext_vector_type(16))) _Float16 v16h;
typedef __attribute__((ext_vector_type(8)))  _Float16 v8h;
typedef __attribute__((ext_vector_type(8)))  float    v8f;

#define D_RES   100
#define DP      112     // 100 padded to 7*16
#define P_PAIRS 128
#define IN_F    128
#define HID_F   64
#define NCLS    10
#define KTOT    20000
#define PITCH   136     // 272B row pitch: 16B-aligned, conflict-free b128 reads

__device__ __forceinline__ v8f wmma_f16(v16h a, v16h b, v8f c) {
  return __builtin_amdgcn_wmma_f32_16x16x32_f16(false, a, false, b,
                                                (short)0, c, false, false);
}
__device__ __forceinline__ v16h cat8(v8h lo, v8h hi) {
  return __builtin_shufflevector(lo, hi, 0, 1, 2, 3, 4, 5, 6, 7,
                                 8, 9, 10, 11, 12, 13, 14, 15);
}
// A fragment (16-bit 16x32): halves 0-7 = K{0..7}+8*hi, 8-15 = K{16..23}+8*hi,
// both contiguous in a row-major [m][k] buffer -> two ds_load_b128.
__device__ __forceinline__ v16h fragA(const _Float16* row, int kc, int hi) {
  v8h lo = *(const v8h*)(row + kc * 32 + hi * 8);
  v8h hh = *(const v8h*)(row + kc * 32 + 16 + hi * 8);
  return cat8(lo, hh);
}
// B fragment (16-bit 32x16): halves 0-7 = K{0..7}+8*hi, 8-15 = K{16..23}+8*hi,
// in a K-contiguous-per-N (transposed) buffer -> two ds_load_b128.
__device__ __forceinline__ v16h fragB(const _Float16* row, int kc, int hi) {
  v8h lo = *(const v8h*)(row + kc * 32 + hi * 8);
  v8h hh = *(const v8h*)(row + kc * 32 + 16 + hi * 8);
  return cat8(lo, hh);
}

// ---------------------------------------------------------------------------
// Kernel 1: node_filt = sigmoid(relu(x@w1 + b1) @ w2 + b2)
// One block = 128 rows (8 waves x 16-row WMMA tiles). K=128, N=64.
// ---------------------------------------------------------------------------
__global__ void __launch_bounds__(256) mlp_filt_kernel(
    const float* __restrict__ x, const float* __restrict__ w1,
    const float* __restrict__ b1, const float* __restrict__ w2,
    const float* __restrict__ b2, float* __restrict__ nf) {
  __shared__ __align__(32) _Float16 xh[128][PITCH];   // [m][k]
  __shared__ __align__(32) _Float16 w1p[HID_F][PITCH];// [n][k] (transposed)
  __shared__ float b1s[HID_F];
  __shared__ float w2s[HID_F];

  const int tid = threadIdx.x;
  const int blockRow = blockIdx.x * 128;

  for (int t = tid; t < 128 * IN_F; t += 256) {
    int r = t >> 7, k = t & 127;
    xh[r][k] = (_Float16)x[(size_t)(blockRow + r) * IN_F + k];
  }
  for (int t = tid; t < IN_F * HID_F; t += 256) {
    int k = t >> 6, n = t & 63;
    w1p[n][k] = (_Float16)w1[t];
  }
  if (tid < HID_F) { b1s[tid] = b1[tid]; w2s[tid] = w2[tid]; }
  __syncthreads();

  const int lane = tid & 31, wave = tid >> 5;
  const int hi = (lane >= 16) ? 1 : 0;
  const int lm = lane & 15;
  const int m0 = wave * 16;

  v8f acc[4];
#pragma unroll
  for (int nt = 0; nt < 4; ++nt)
#pragma unroll
    for (int r = 0; r < 8; ++r) acc[nt][r] = 0.0f;

  const _Float16* arow = &xh[m0 + lm][0];
#pragma unroll
  for (int kc = 0; kc < 4; ++kc) {
    v16h a = fragA(arow, kc, hi);
#pragma unroll
    for (int nt = 0; nt < 4; ++nt) {
      v16h bf = fragB(&w1p[nt * 16 + lm][0], kc, hi);
      acc[nt] = wmma_f16(a, bf, acc[nt]);
    }
  }

  // Fused epilogue: relu(+b1), dot with w2 across N via wave32 butterfly.
  float part[8];
#pragma unroll
  for (int r = 0; r < 8; ++r) part[r] = 0.0f;
#pragma unroll
  for (int nt = 0; nt < 4; ++nt) {
    float bb = b1s[nt * 16 + lm];
    float ww = w2s[nt * 16 + lm];
#pragma unroll
    for (int r = 0; r < 8; ++r) {
      float h = acc[nt][r] + bb;
      h = h > 0.0f ? h : 0.0f;
      part[r] += h * ww;
    }
  }
#pragma unroll
  for (int mask = 1; mask < 16; mask <<= 1)
#pragma unroll
    for (int r = 0; r < 8; ++r) part[r] += __shfl_xor(part[r], mask, 32);

  if (lm == 0) {
    float bias2 = b2[0];
#pragma unroll
    for (int r = 0; r < 8; ++r) {
      int row = blockRow + m0 + r + hi * 8;
      float z = part[r] + bias2;
      nf[row] = 1.0f / (1.0f + __expf(-z));
    }
  }
}

// ---------------------------------------------------------------------------
// Kernel 2: persistence image as rank-128 outer product GEMM + max-normalize.
// One block per (graph, channel). im[i,j] = sum_p expA[p,i]*expB[p,j].
// ---------------------------------------------------------------------------
__global__ void __launch_bounds__(256) pers_image_kernel(
    const int* __restrict__ pair_idx0, const int* __restrict__ pair_idx1,
    const float* __restrict__ nf, float* __restrict__ imgs) {
  __shared__ __align__(32) _Float16 Ab[DP][PITCH];  // [i][p] birth Gaussians
  __shared__ __align__(32) _Float16 Bt[DP][PITCH];  // [j][p] pers Gaussians
  __shared__ float birth[P_PAIRS];
  __shared__ float pers[P_PAIRS];
  __shared__ float wmaxArr[8];
  __shared__ float gmaxSh;

  const int tid = threadIdx.x;
  const int b = blockIdx.x >> 1;
  const int c = blockIdx.x & 1;
  const int* idx = (c == 0 ? pair_idx0 : pair_idx1) + (size_t)b * P_PAIRS * 2;

  if (tid < P_PAIRS) {
    int i0 = idx[tid * 2], i1 = idx[tid * 2 + 1];
    float f0 = nf[i0], f1 = nf[i1];
    birth[tid] = f0;
    pers[tid] = f1 - f0;
  }
  __syncthreads();

  // 2*VAR = 1.0 -> exp(-d^2); 1/(2*pi*VAR) cancels under max-normalization.
  for (int t = tid; t < DP * P_PAIRS; t += 256) {
    int i = t >> 7, p = t & 127;
    float v = 0.0f;
    if (i < D_RES) { float d = birth[p] - (float)i * 0.01f; v = __expf(-d * d); }
    Ab[i][p] = (_Float16)v;
  }
  for (int t = tid; t < DP * P_PAIRS; t += 256) {
    int j = t >> 7, p = t & 127;
    float v = 0.0f;
    if (j < D_RES) { float d = pers[p] - (float)j * 0.01f; v = __expf(-d * d); }
    Bt[j][p] = (_Float16)v;
  }
  __syncthreads();

  const int lane = tid & 31, wave = tid >> 5;
  const int hi = (lane >= 16) ? 1 : 0;
  const int lm = lane & 15;
  float* img = imgs + (size_t)blockIdx.x * D_RES * D_RES;
  float wmax = 0.0f;

  for (int t = wave; t < 49; t += 8) {  // 7x7 tiles of 16x16
    int mt = t / 7, nt = t - mt * 7;
    const _Float16* arow = &Ab[mt * 16 + lm][0];
    const _Float16* brow = &Bt[nt * 16 + lm][0];
    v8f acc;
#pragma unroll
    for (int r = 0; r < 8; ++r) acc[r] = 0.0f;
#pragma unroll
    for (int kc = 0; kc < 4; ++kc)
      acc = wmma_f16(fragA(arow, kc, hi), fragB(brow, kc, hi), acc);

    int n = nt * 16 + lm;
    if (n < D_RES) {
#pragma unroll
      for (int r = 0; r < 8; ++r) {
        int m = mt * 16 + r + hi * 8;
        if (m < D_RES) {
          float v = acc[r];
          img[m * D_RES + n] = v;
          wmax = wmax > v ? wmax : v;
        }
      }
    }
  }
#pragma unroll
  for (int mask = 1; mask < 32; mask <<= 1) {
    float o = __shfl_xor(wmax, mask, 32);
    wmax = wmax > o ? wmax : o;
  }
  if (lane == 0) wmaxArr[wave] = wmax;
  __threadfence();
  __syncthreads();
  if (tid == 0) {
    float g = wmaxArr[0];
    for (int i = 1; i < 8; ++i) g = g > wmaxArr[i] ? g : wmaxArr[i];
    gmaxSh = g;
  }
  __syncthreads();
  float inv = 1.0f / gmaxSh;  // sums of exp() are strictly positive
  for (int t = tid; t < D_RES * D_RES; t += 256) img[t] *= inv;
}

// ---------------------------------------------------------------------------
// Kernel 3: conv1 (2->16, 5x5, pad2) + relu + 2x2 maxpool. Direct VALU.
// ---------------------------------------------------------------------------
__global__ void __launch_bounds__(256) conv1_pool_kernel(
    const float* __restrict__ imgs, const float* __restrict__ w,
    const float* __restrict__ bias, float* __restrict__ out) {
  int o = blockIdx.x * 256 + threadIdx.x;
  if (o >= 32 * 16 * 50 * 50) return;
  int px = o % 50;
  int py = (o / 50) % 50;
  int ch = (o / 2500) % 16;
  int b = o / 40000;
  float bv = bias[ch];
  float best = 0.0f;  // relu output is >= 0
#pragma unroll
  for (int dy = 0; dy < 2; ++dy)
#pragma unroll
    for (int dx = 0; dx < 2; ++dx) {
      int oy = py * 2 + dy, ox = px * 2 + dx;
      float sum = bv;
      for (int ci = 0; ci < 2; ++ci) {
        const float* ip = imgs + (size_t)(b * 2 + ci) * 10000;
        const float* wp = w + (ch * 2 + ci) * 25;
#pragma unroll
        for (int ky = 0; ky < 5; ++ky) {
          int iy = oy + ky - 2;
          if (iy < 0 || iy >= 100) continue;
#pragma unroll
          for (int kx = 0; kx < 5; ++kx) {
            int ix = ox + kx - 2;
            if (ix < 0 || ix >= 100) continue;
            sum += ip[iy * 100 + ix] * wp[ky * 5 + kx];
          }
        }
      }
      sum = sum > 0.0f ? sum : 0.0f;
      best = best > sum ? best : sum;
    }
  out[o] = best;
}

// ---------------------------------------------------------------------------
// Kernel 4: conv2 (16->32, 5x5, pad2) + relu + 2x2 maxpool. Direct VALU.
// ---------------------------------------------------------------------------
__global__ void __launch_bounds__(256) conv2_pool_kernel(
    const float* __restrict__ inp, const float* __restrict__ w,
    const float* __restrict__ bias, float* __restrict__ out) {
  int o = blockIdx.x * 256 + threadIdx.x;
  if (o >= 32 * 32 * 25 * 25) return;
  int px = o % 25;
  int py = (o / 25) % 25;
  int ch = (o / 625) % 32;
  int b = o / 20000;
  float bv = bias[ch];
  float best = 0.0f;
  for (int dy = 0; dy < 2; ++dy)
    for (int dx = 0; dx < 2; ++dx) {
      int oy = py * 2 + dy, ox = px * 2 + dx;
      float sum = bv;
      for (int ci = 0; ci < 16; ++ci) {
        const float* ip = inp + (size_t)(b * 16 + ci) * 2500;
        const float* wp = w + (ch * 16 + ci) * 25;
#pragma unroll
        for (int ky = 0; ky < 5; ++ky) {
          int iy = oy + ky - 2;
          if (iy < 0 || iy >= 50) continue;
#pragma unroll
          for (int kx = 0; kx < 5; ++kx) {
            int ix = ox + kx - 2;
            if (ix < 0 || ix >= 50) continue;
            sum += ip[iy * 50 + ix] * wp[ky * 5 + kx];
          }
        }
      }
      sum = sum > 0.0f ? sum : 0.0f;
      best = best > sum ? best : sum;
    }
  out[o] = best;
}

// ---------------------------------------------------------------------------
// Kernel 5: y_hat[32,10] = yflat[32,20000] @ out_w.T + out_b via WMMA.
// 8 waves: wave = mtile(2) x kslice(4); K = 625 chunks of 32; LDS reduce.
// Fragments loaded straight from global as float4 (b128) + cvt to f16.
// ---------------------------------------------------------------------------
__global__ void __launch_bounds__(256) head_linear_kernel(
    const float* __restrict__ yflat, const float* __restrict__ out_w,
    const float* __restrict__ out_b, float* __restrict__ yhat) {
  __shared__ float red[8][256];
  const int tid = threadIdx.x;
  const int lane = tid & 31, wave = tid >> 5;
  const int hi = (lane >= 16) ? 1 : 0;
  const int lm = lane & 15;
  const int mt = wave & 1;
  const int ks = wave >> 1;
  int c0 = ks * 157;
  int c1 = c0 + 157;
  if (c1 > 625) c1 = 625;
  const int m = mt * 16 + lm;
  const float* arow = yflat + (size_t)m * KTOT;
  const float* brow = out_w + (size_t)(lm < NCLS ? lm : 0) * KTOT;
  const bool bvalid = (lm < NCLS);

  v8f acc;
#pragma unroll
  for (int r = 0; r < 8; ++r) acc[r] = 0.0f;

  for (int ck = c0; ck < c1; ++ck) {
    int kbase = ck * 32;
    // A: halves 0-7 at K=kbase+8hi..+7, halves 8-15 at K=kbase+16+8hi..+7
    int ka0 = kbase + hi * 8;
    int ka1 = kbase + 16 + hi * 8;
    float4 a0 = *(const float4*)(arow + ka0);
    float4 a1 = *(const float4*)(arow + ka0 + 4);
    float4 a2 = *(const float4*)(arow + ka1);
    float4 a3 = *(const float4*)(arow + ka1 + 4);
    // B: halves 0-7 at K=kbase+8hi..+7, halves 8-15 at K=kbase+16+8hi..+7 (zero for padded classes)
    int kb0 = kbase + hi * 8;
    float4 b0 = make_float4(0.f, 0.f, 0.f, 0.f), b1 = b0, b2 = b0, b3 = b0;
    if (bvalid) {
      b0 = *(const float4*)(brow + kb0);
      b1 = *(const float4*)(brow + kb0 + 4);
      b2 = *(const float4*)(brow + kb0 + 16);
      b3 = *(const float4*)(brow + kb0 + 20);
    }
    v16h a, bf;
    a[0] = (_Float16)a0.x;  a[1] = (_Float16)a0.y;
    a[2] = (_Float16)a0.z;  a[3] = (_Float16)a0.w;
    a[4] = (_Float16)a1.x;  a[5] = (_Float16)a1.y;
    a[6] = (_Float16)a1.z;  a[7] = (_Float16)a1.w;
    a[8] = (_Float16)a2.x;  a[9] = (_Float16)a2.y;
    a[10] = (_Float16)a2.z; a[11] = (_Float16)a2.w;
    a[12] = (_Float16)a3.x; a[13] = (_Float16)a3.y;
    a[14] = (_Float16)a3.z; a[15] = (_Float16)a3.w;
    bf[0] = (_Float16)b0.x;  bf[1] = (_Float16)b0.y;
    bf[2] = (_Float16)b0.z;  bf[3] = (_Float16)b0.w;
    bf[4] = (_Float16)b1.x;  bf[5] = (_Float16)b1.y;
    bf[6] = (_Float16)b1.z;  bf[7] = (_Float16)b1.w;
    bf[8] = (_Float16)b2.x;  bf[9] = (_Float16)b2.y;
    bf[10] = (_Float16)b2.z; bf[11] = (_Float16)b2.w;
    bf[12] = (_Float16)b3.x; bf[13] = (_Float16)b3.y;
    bf[14] = (_Float16)b3.z; bf[15] = (_Float16)b3.w;
    acc = wmma_f16(a, bf, acc);
  }
#pragma unroll
  for (int r = 0; r < 8; ++r) red[wave][lane * 8 + r] = acc[r];
  __syncthreads();

  for (int mtt = 0; mtt < 2; ++mtt) {
    int j = tid;
    float s = red[mtt][j] + red[mtt + 2][j] + red[mtt + 4][j] + red[mtt + 6][j];
    int L = j >> 3, r = j & 7;
    int hh = (L >= 16) ? 1 : 0;
    int mm = mtt * 16 + r + hh * 8;
    int nn = L & 15;
    if (nn < NCLS) yhat[mm * NCLS + nn] = s + out_b[nn];
  }
}

// ---------------------------------------------------------------------------
extern "C" void kernel_launch(void* const* d_in, const int* in_sizes, int n_in,
                              void* d_out, int out_size, void* d_ws,
                              size_t ws_size, hipStream_t stream) {
  const float* x       = (const float*)d_in[0];
  const int*   pair0   = (const int*)d_in[1];
  const int*   pair1   = (const int*)d_in[2];
  const float* w1      = (const float*)d_in[3];
  const float* b1      = (const float*)d_in[4];
  const float* w2      = (const float*)d_in[5];
  const float* b2      = (const float*)d_in[6];
  const float* conv1_w = (const float*)d_in[7];
  const float* conv1_b = (const float*)d_in[8];
  const float* conv2_w = (const float*)d_in[9];
  const float* conv2_b = (const float*)d_in[10];
  const float* out_w   = (const float*)d_in[11];
  const float* out_b   = (const float*)d_in[12];

  float* yhat = (float*)d_out;            // [32,10]
  float* nf   = (float*)d_out + 320;      // [16000]

  float* ws    = (float*)d_ws;
  float* imgs  = ws;                      // 32*2*100*100  = 640000 f
  float* pool1 = ws + 640000;             // 32*16*50*50   = 1280000 f
  float* pool2 = ws + 640000 + 1280000;   // 32*32*25*25   = 640000 f

  mlp_filt_kernel<<<125, 256, 0, stream>>>(x, w1, b1, w2, b2, nf);
  pers_image_kernel<<<64, 256, 0, stream>>>(pair0, pair1, nf, imgs);
  conv1_pool_kernel<<<(32 * 16 * 50 * 50 + 255) / 256, 256, 0, stream>>>(
      imgs, conv1_w, conv1_b, pool1);
  conv2_pool_kernel<<<(32 * 32 * 25 * 25 + 255) / 256, 256, 0, stream>>>(
      pool1, conv2_w, conv2_b, pool2);
  head_linear_kernel<<<1, 256, 0, stream>>>(pool2, out_w, out_b, yhat);
}
